// IMMA_54906861912219
// MI455X (gfx1250) — compile-verified, matrix operand order checked by reading, not run on hardware
//
#include <hip/hip_runtime.h>
#include <hip/hip_bf16.h>

// ---------------------------------------------------------------------------
// NRI encoder + GRU decoder rollout for MI455X (gfx1250, wave32, WMMA).
// All matmuls run through v_wmma_f32_16x16x32_f16 (f16 in, f32 accum); weight
// tiles are staged into LDS with CDNA5 async loads
// (global_load_async_to_lds_b128, ASYNCcnt) so the copy runs under the WMMAs;
// activation tiles are loaded with true global_load_b128 (addrspace(1)).
// Assumed d_in flattening: top-level dict insertion order, nested pytrees in
// JAX leaf order (dict keys sorted alphabetically, lists by index):
//   0: x
//   1+26k+j : enc_params[k]  (j: 0 fc_out.W, 1 fc_out.b,
//             2..7  mlp1{bn_b,bn_g,fc1.W,fc1.b,fc2.W,fc2.b},
//             8..13 mlp2{...}, 14..19 mlp3{...}, 20..25 mlp4{...})
//   53..75 : dec_params sorted: hidden_h.W, hidden_i.W, hidden_r.W,
//            input_i.{W,b}, input_n.{W,b}, input_r.{W,b},
//            msg_fc1[0].{W,b}, msg_fc1[1].{W,b}, msg_fc2[0].{W,b}, msg_fc2[1].{W,b},
//            out_fc1.{W,b}, out_fc2.{W,b}, out_fc3.{W,b}
//   76,77: trans.{W,b}  78,79: head.{W,b}  80: recv_idx  81: send_idx
//   82: burn_in_length  83: rollout_length
// ---------------------------------------------------------------------------

#define BBATCH 64
#define NNODE  64
#define TTIME  24
#define DDIM   8
#define HDIM   128
#define EEDGE  4032          // N*(N-1); recv[e] == e/63, groups of 63
#define RROLL  16
#define MBE    (BBATCH*EEDGE)   // 258048 rows (edge GEMMs) == 2016 * 128
#define MBN    (BBATCH*NNODE)   // 4096 rows (node GEMMs)   == 32 * 128

typedef __attribute__((ext_vector_type(16))) _Float16 v16h;
typedef __attribute__((ext_vector_type(8)))  _Float16 v8h;
typedef __attribute__((ext_vector_type(8)))  float    v8f;
typedef __attribute__((ext_vector_type(4)))  float    f4raw;   // builtin vector (no class ctor)

// Explicit global-addrspace pointer so A-tile loads lower to global_load_b128
// (flat loads would also tick DScnt and serialize against LDS fragment waits).
// Must point at a builtin vector type: HIP's float4 class ctor cannot bind an
// addrspace(1) reference.
typedef const f4raw __attribute__((address_space(1)))* gcf4;

struct GemmSeg {
    const float* ptr;   // source activations, row pitch = ld floats
    const int*   idx;   // non-null: edge gather, srcRow = b*NNODE + idx[e]
    int          ld;
};

enum { ACT_NONE = 0, ACT_ELU, ACT_TANH, ACT_RELU, ACT_SIGMOID };

template <int ACT>
__device__ __forceinline__ float act_apply(float x) {
    if (ACT == ACT_ELU)     return x > 0.f ? x : (__expf(x) - 1.f);
    if (ACT == ACT_TANH)    return tanhf(x);
    if (ACT == ACT_RELU)    return x > 0.f ? x : 0.f;
    if (ACT == ACT_SIGMOID) return 1.f / (1.f + __expf(-x));
    return x;
}

#define LDST 40   // LDS row pitch in halves (32 data + 8 pad; rows stay 16B aligned)

// CDNA5 async copy: 16 bytes global -> LDS, tracked by ASYNCcnt (no VGPR data).
__device__ __forceinline__ void async_g2l_b128(const void* gsrc, void* ldst) {
    unsigned loff = (unsigned)(uintptr_t)ldst;   // low 32 bits = LDS byte offset
    asm volatile("global_load_async_to_lds_b128 %0, %1, off"
                 :: "v"(loff), "v"(gsrc) : "memory");
}
__device__ __forceinline__ void wait_asynccnt0() {
    asm volatile("s_wait_asynccnt 0x0" ::: "memory");
}

// ---------------------------------------------------------------------------
// Tiled GEMM: C[M,128] = epilogue( A[M,Kin] @ Wt^T + bias )
//   * A described by up to 3 segments (fused gather-concat), f32 -> f16 in LDS
//   * Wt pre-converted f16, layout [128][Kin] (W transposed); staged to LDS
//     via global_load_async_to_lds_b128 (fire-and-forget, ASYNCcnt)
//   * Block: 256 threads = 8 waves; tile 128x128; K chunk 32
//   * Double-buffered LDS; next chunk's A loads + async W copies issue before
//     the WMMA sequence; one barrier per chunk.
//   * M must be a multiple of 128 (true for all call sites) -> no tail guards.
// ---------------------------------------------------------------------------
template <int ACT>
__global__ __launch_bounds__(256)
void gemm128_wmma(GemmSeg s0, GemmSeg s1, GemmSeg s2,
                  int kStart1, int kStart2,
                  const _Float16* __restrict__ Wt,
                  const float* __restrict__ bias,
                  const float* __restrict__ addend,
                  const float* __restrict__ rowScale, int rsStride, int rsOff, float rsFactor,
                  int accumulate,
                  float* __restrict__ C,
                  int M, int Kin)
{
    (void)M;
    __shared__ _Float16 ldsA[2][128 * LDST];
    __shared__ _Float16 ldsB[2][128 * LDST];

    const int tid     = threadIdx.x;
    const int wave    = tid >> 5;
    const int lane    = tid & 31;
    const int l15     = lane & 15;
    const int rowBase = blockIdx.x * 128;

    // ---- per-thread staging coordinates (chunk-invariant) ----
    const int r    = tid >> 1;       // tile row (A) / weight output row (B)
    const int half = tid & 1;        // which 16-wide half of the 32-wide chunk
    const int gr   = rowBase + r;
    const int bI   = gr / EEDGE;     // batch (only meaningful for edge GEMMs)
    const int eI   = gr - bI * EEDGE;
    const float* rp0 = s0.ptr + (size_t)(s0.idx ? (bI * NNODE + s0.idx[eI]) : gr) * s0.ld;
    const float* rp1 = s1.ptr ? s1.ptr + (size_t)(s1.idx ? (bI * NNODE + s1.idx[eI]) : gr) * s1.ld
                              : rp0;
    const float* rp2 = s2.ptr ? s2.ptr + (size_t)(s2.idx ? (bI * NNODE + s2.idx[eI]) : gr) * s2.ld
                              : rp0;
    const _Float16* WtRow = Wt + (size_t)r * Kin;

    auto a_src = [&](int kcb) -> gcf4 {
        const float* rp = (kcb >= kStart2) ? rp2 : (kcb >= kStart1 ? rp1 : rp0);
        const int    k0 = (kcb >= kStart2) ? kStart2 : (kcb >= kStart1 ? kStart1 : 0);
        return (gcf4)(uintptr_t)(rp + (kcb - k0) + half * 16);
    };
    auto w_stage = [&](int kcb, int buf) {
        const _Float16* src = WtRow + kcb + half * 16;
        _Float16*       dst = &ldsB[buf][r * LDST + half * 16];
        async_g2l_b128(src, dst);
        async_g2l_b128(src + 8, dst + 8);
    };
    auto a_store = [&](int buf, f4raw a0, f4raw a1, f4raw a2, f4raw a3) {
        _Float16* da = &ldsA[buf][r * LDST + half * 16];
        da[0]=(_Float16)a0[0]; da[1]=(_Float16)a0[1]; da[2]=(_Float16)a0[2]; da[3]=(_Float16)a0[3];
        da[4]=(_Float16)a1[0]; da[5]=(_Float16)a1[1]; da[6]=(_Float16)a1[2]; da[7]=(_Float16)a1[3];
        da[8]=(_Float16)a2[0]; da[9]=(_Float16)a2[1]; da[10]=(_Float16)a2[2]; da[11]=(_Float16)a2[3];
        da[12]=(_Float16)a3[0]; da[13]=(_Float16)a3[1]; da[14]=(_Float16)a3[2]; da[15]=(_Float16)a3[3];
    };

    // ---- fragment coordinates ----
    const int mrow = (wave << 4) + l15;
    const int kb   = (lane >> 4) << 3;   // 0 or 8 halves (ISA 16-bit A 16x32 layout)

    v8f acc[8];
#pragma unroll
    for (int ct = 0; ct < 8; ++ct)
#pragma unroll
        for (int i = 0; i < 8; ++i) acc[ct][i] = 0.f;

    const int nChunks = Kin >> 5;

    // ---- prologue: stage chunk 0 into buffer 0 ----
    {
        w_stage(0, 0);
        gcf4 s4 = a_src(0);
        f4raw a0 = s4[0], a1 = s4[1], a2 = s4[2], a3 = s4[3];
        a_store(0, a0, a1, a2, a3);
    }
    wait_asynccnt0();
    __syncthreads();

    for (int kc = 0; kc < nChunks; ++kc) {
        const int  buf  = kc & 1;
        const bool more = (kc + 1) < nChunks;

        // ---- issue next chunk's loads: A to VGPRs (no wait), W async to LDS ----
        f4raw a0, a1, a2, a3;
        if (more) {
            const int kcb = (kc + 1) << 5;
            w_stage(kcb, buf ^ 1);
            gcf4 s4 = a_src(kcb);
            a0 = s4[0]; a1 = s4[1]; a2 = s4[2]; a3 = s4[3];
        }

        // ---- compute current chunk: batch all LDS fragment loads, then WMMAs ----
        {
            const _Float16* lA = ldsA[buf];
            const _Float16* lB = ldsB[buf];
            v8h alo = *(const v8h*)&lA[mrow * LDST + kb];
            v8h ahi = *(const v8h*)&lA[mrow * LDST + kb + 16];
            v16h afrag;
#pragma unroll
            for (int i = 0; i < 8; ++i) { afrag[i] = alo[i]; afrag[i + 8] = ahi[i]; }
            v8h bl[8], bh[8];
#pragma unroll
            for (int ct = 0; ct < 8; ++ct) {
                const int nrow = (ct << 4) + l15;
                bl[ct] = *(const v8h*)&lB[nrow * LDST + kb];
                bh[ct] = *(const v8h*)&lB[nrow * LDST + kb + 16];
            }
#pragma unroll
            for (int ct = 0; ct < 8; ++ct) {
                v16h bfrag;
#pragma unroll
                for (int i = 0; i < 8; ++i) { bfrag[i] = bl[ct][i]; bfrag[i + 8] = bh[ct][i]; }
                acc[ct] = __builtin_amdgcn_wmma_f32_16x16x32_f16(
                    false, afrag, false, bfrag, (short)0, acc[ct], false, false);
            }
        }

        // ---- convert + store next A chunk into the other buffer ----
        if (more) a_store(buf ^ 1, a0, a1, a2, a3);
        wait_asynccnt0();
        __syncthreads();
    }

    // ---- epilogue: C/D layout VGPR r -> row (lane<16 ? r : 8+r), col = lane%16 ----
    const int rOff = (lane >> 4) << 3;
#pragma unroll
    for (int ct = 0; ct < 8; ++ct) {
        const int col = (ct << 4) + l15;
        const float bv = bias ? bias[col] : 0.f;
#pragma unroll
        for (int rr = 0; rr < 8; ++rr) {
            const int row = rowBase + (wave << 4) + rOff + rr;
            float v = acc[ct][rr] + bv;
            if (addend) v += addend[(size_t)row * HDIM + col];
            v = act_apply<ACT>(v);
            if (rowScale) v *= rowScale[(size_t)row * rsStride + rsOff] * rsFactor;
            const size_t o = (size_t)row * HDIM + col;
            if (accumulate) C[o] += v; else C[o] = v;
        }
    }
}

// ---------------------------------------------------------------------------
// Small helper kernels
// ---------------------------------------------------------------------------
__global__ void wt_convert_kernel(const float* __restrict__ W, _Float16* __restrict__ Wt,
                                  int Kin, int ldK, int kOff) {
    int i = blockIdx.x * blockDim.x + threadIdx.x;   // over Kin*128
    if (i >= Kin * 128) return;
    int k = i >> 7, n = i & 127;
    Wt[(size_t)n * ldK + kOff + k] = (_Float16)W[i]; // W is (Kin,128) row-major
}

__global__ void zero_kernel(float* p, int n) {
    int i = blockIdx.x * blockDim.x + threadIdx.x;
    if (i < n) p[i] = 0.f;
}

__global__ void flat_kernel(const float* __restrict__ x, float* __restrict__ flat) {
    int i = blockIdx.x * blockDim.x + threadIdx.x;   // over B*N*T*D
    if (i >= BBATCH * NNODE * TTIME * DDIM) return;
    int d = i & 7;
    int t = (i >> 3) % TTIME;
    int n = (i / (TTIME * DDIM)) % NNODE;
    int b = i / (NNODE * TTIME * DDIM);
    flat[((b * NNODE + n) * (TTIME * DDIM)) + t * DDIM + d] =
        x[((b * TTIME + t) * (NNODE * DDIM)) + n * DDIM + d];
}

__global__ void bnstats_kernel(const float* __restrict__ X, float* __restrict__ st, int M) {
    int c = threadIdx.x;                 // 128 channels
    int r0 = blockIdx.x * 512;
    float s = 0.f, ss = 0.f;
    for (int i = 0; i < 512; ++i) {
        int r = r0 + i;
        if (r < M) { float v = X[(size_t)r * HDIM + c]; s += v; ss += v * v; }
    }
    atomicAdd(&st[c], s);
    atomicAdd(&st[128 + c], ss);
}

__global__ void bnapply_kernel(float* __restrict__ X, const float* __restrict__ st,
                               const float* __restrict__ g, const float* __restrict__ b, int M) {
    int i = blockIdx.x * blockDim.x + threadIdx.x;   // over M*128
    if (i >= M * HDIM) return;
    int c = i & 127;
    float m = st[c] / (float)M;
    float v = st[128 + c] / (float)M - m * m;
    X[i] = (X[i] - m) * rsqrtf(v + 1e-5f) * g[c] + b[c];
}

__global__ void fcout_kernel(const float* __restrict__ e2, const float* __restrict__ w,
                             const float* __restrict__ b, float* __restrict__ logits) {
    int r = blockIdx.x * blockDim.x + threadIdx.x;   // over B*E
    if (r >= MBE) return;
    float s = b[0];
    const float4* row = (const float4*)(e2 + (size_t)r * HDIM);
    const float4* w4  = (const float4*)w;
    for (int c = 0; c < HDIM / 4; ++c) {
        float4 a = row[c], q = w4[c];
        s += a.x * q.x + a.y * q.y + a.z * q.z + a.w * q.w;
    }
    logits[r] = s;
}

__global__ void softmax63_kernel(const float* __restrict__ logits, float* __restrict__ rel, int k) {
    int g = blockIdx.x * blockDim.x + threadIdx.x;   // over B*N groups of 63
    if (g >= MBN) return;
    int base = g * 63;                               // b*4032 + n*63
    float mx = -1e30f;
    for (int j = 0; j < 63; ++j) mx = fmaxf(mx, logits[base + j]);
    float s = 0.f;
    for (int j = 0; j < 63; ++j) s += __expf(logits[base + j] - mx);
    float inv = 1.f / s;
    for (int j = 0; j < 63; ++j)
        rel[(size_t)(base + j) * 2 + k] = __expf(logits[base + j] - mx) * inv;
}

__global__ void agg_kernel(const float* __restrict__ edges, float* __restrict__ outn, float scale) {
    int i = blockIdx.x * blockDim.x + threadIdx.x;   // over B*N*H
    if (i >= MBN * HDIM) return;
    int c = i & 127;
    int g = i >> 7;                                  // b*64 + n; edges for node n: [g*63, g*63+63)
    int base = g * 63;
    float s = 0.f;
    for (int j = 0; j < 63; ++j) s += edges[(size_t)(base + j) * HDIM + c];
    outn[i] = s * scale;
}

__global__ void gru_kernel(float* __restrict__ hidden, const float* __restrict__ rb,
                           const float* __restrict__ ib, const float* __restrict__ n1,
                           const float* __restrict__ n2) {
    int i = blockIdx.x * blockDim.x + threadIdx.x;   // over B*N*H
    if (i >= MBN * HDIM) return;
    float nn = tanhf(n1[i] + rb[i] * n2[i]);
    float ii = ib[i];
    hidden[i] = (1.f - ii) * nn + ii * hidden[i];
}

__global__ void initacc_kernel(const float* __restrict__ x, float* __restrict__ accD) {
    int i = blockIdx.x * blockDim.x + threadIdx.x;   // over B*N*D
    if (i >= MBN * DDIM) return;
    int d = i & 7, n = (i >> 3) & 63, b = i >> 9;
    accD[i] = x[((b * TTIME + (TTIME - 1)) * (NNODE * DDIM)) + n * DDIM + d];
}

__global__ void head_out_kernel(const float* __restrict__ pred, const float* __restrict__ hW,
                                const float* __restrict__ hb, float* __restrict__ accD,
                                float* __restrict__ out, int step) {
    int i = blockIdx.x * blockDim.x + threadIdx.x;   // over B*N*D
    if (i >= MBN * DDIM) return;
    int d = i & 7;
    int nrow = i >> 3;                               // b*64+n
    float s = hb[d];
    const float* prow = pred + (size_t)nrow * HDIM;
    for (int c = 0; c < HDIM; ++c) s += prow[c] * hW[c * DDIM + d];
    float a = accD[i] + s;
    accD[i] = a;
    int b = nrow >> 6, n = nrow & 63;
    out[((size_t)b * RROLL + step) * (NNODE * DDIM) + n * DDIM + d] = a;
}

// ---------------------------------------------------------------------------
// Host side
// ---------------------------------------------------------------------------
static void launch_gemm(int act, hipStream_t st, int M, int Kin,
                        GemmSeg s0, GemmSeg s1, GemmSeg s2, int k1, int k2,
                        const _Float16* Wt, const float* bias, float* C,
                        const float* addend = nullptr,
                        const float* rowScale = nullptr, int rsStride = 0, int rsOff = 0,
                        float rsFactor = 1.f, int accum = 0) {
    dim3 g(M / 128), b(256);
    switch (act) {
    case ACT_NONE:    gemm128_wmma<ACT_NONE><<<g, b, 0, st>>>(s0, s1, s2, k1, k2, Wt, bias, addend, rowScale, rsStride, rsOff, rsFactor, accum, C, M, Kin); break;
    case ACT_ELU:     gemm128_wmma<ACT_ELU><<<g, b, 0, st>>>(s0, s1, s2, k1, k2, Wt, bias, addend, rowScale, rsStride, rsOff, rsFactor, accum, C, M, Kin); break;
    case ACT_TANH:    gemm128_wmma<ACT_TANH><<<g, b, 0, st>>>(s0, s1, s2, k1, k2, Wt, bias, addend, rowScale, rsStride, rsOff, rsFactor, accum, C, M, Kin); break;
    case ACT_RELU:    gemm128_wmma<ACT_RELU><<<g, b, 0, st>>>(s0, s1, s2, k1, k2, Wt, bias, addend, rowScale, rsStride, rsOff, rsFactor, accum, C, M, Kin); break;
    default:          gemm128_wmma<ACT_SIGMOID><<<g, b, 0, st>>>(s0, s1, s2, k1, k2, Wt, bias, addend, rowScale, rsStride, rsOff, rsFactor, accum, C, M, Kin); break;
    }
}

extern "C" void kernel_launch(void* const* d_in, const int* in_sizes, int n_in,
                              void* d_out, int out_size, void* d_ws, size_t ws_size,
                              hipStream_t stream) {
    (void)in_sizes; (void)n_in; (void)out_size; (void)ws_size;
    auto F = [&](int i) -> const float* { return (const float*)d_in[i]; };
    const float* X    = F(0);
    const int*   recv = (const int*)d_in[80];
    const int*   send = (const int*)d_in[81];
    float*       out  = (float*)d_out;

    // ---- workspace carve (f32 region then f16 weight region) ----
    float* f = (float*)d_ws;
    size_t o = 0;
    auto A = [&](size_t n) -> float* { float* p = f + o; o += (n + 3) & ~(size_t)3; return p; };
    float* flat  = A((size_t)MBN * TTIME * DDIM);
    float* nodeA = A((size_t)MBN * HDIM);
    float* nodeB = A((size_t)MBN * HDIM);
    float* insB  = A((size_t)MBN * HDIM);
    float* hid   = A((size_t)MBN * HDIM);
    float* aggB  = A((size_t)MBN * HDIM);
    float* rbuf  = A((size_t)MBN * HDIM);
    float* ibuf  = A((size_t)MBN * HDIM);
    float* n1b   = A((size_t)MBN * HDIM);
    float* n2b   = A((size_t)MBN * HDIM);
    float* p1b   = A((size_t)MBN * HDIM);
    float* p2b   = A((size_t)MBN * HDIM);
    float* accD  = A((size_t)MBN * DDIM);
    float* rel   = A((size_t)MBE * 2);
    float* logit = A((size_t)MBE);
    float* stats = A(256);
    float* EA    = A((size_t)MBE * HDIM);
    float* EB    = A((size_t)MBE * HDIM);

    _Float16* hreg = (_Float16*)(f + o);
    size_t ho = 0;
    auto HA = [&](size_t n) -> _Float16* { _Float16* p = hreg + ho; ho += (n + 7) & ~(size_t)7; return p; };
    _Float16 *Wm1f1[2], *Wm1f2[2], *Wm2f1[2], *Wm2f2[2], *Wm3f1[2], *Wm3f2[2], *Wm4f1[2], *Wm4f2[2];
    for (int k = 0; k < 2; ++k) {
        Wm1f1[k] = HA(128 * 192); Wm1f2[k] = HA(128 * 128);
        Wm2f1[k] = HA(128 * 256); Wm2f2[k] = HA(128 * 128);
        Wm3f1[k] = HA(128 * 128); Wm3f2[k] = HA(128 * 128);
        Wm4f1[k] = HA(128 * 384); Wm4f2[k] = HA(128 * 128);
    }
    _Float16 *Wmsg1[2], *Wmsg2[2];
    Wmsg1[0] = HA(128 * 256); Wmsg1[1] = HA(128 * 256);
    Wmsg2[0] = HA(128 * 128); Wmsg2[1] = HA(128 * 128);
    _Float16* Wr  = HA(128 * 256);
    _Float16* Wi  = HA(128 * 256);
    _Float16* Wn  = HA(128 * 128);
    _Float16* Whh = HA(128 * 128);
    _Float16* Wo1 = HA(128 * 128);
    _Float16* Wo2 = HA(128 * 128);
    _Float16* Wo3 = HA(128 * 128);
    _Float16* Wtr = HA(128 * 128);

    auto conv = [&](const float* W, _Float16* Wt, int Kin, int ldK, int kOff) {
        int n = Kin * 128;
        wt_convert_kernel<<<(n + 255) / 256, 256, 0, stream>>>(W, Wt, Kin, ldK, kOff);
    };
    auto EIx = [&](int k, int j) { return 1 + k * 26 + j; };

    // ---- weight conversion (f32 -> f16 transposed [128][Kin]) ----
    for (int k = 0; k < 2; ++k) {
        conv(F(EIx(k, 4)),  Wm1f1[k], 192, 192, 0);
        conv(F(EIx(k, 6)),  Wm1f2[k], 128, 128, 0);
        conv(F(EIx(k, 10)), Wm2f1[k], 256, 256, 0);
        conv(F(EIx(k, 12)), Wm2f2[k], 128, 128, 0);
        conv(F(EIx(k, 16)), Wm3f1[k], 128, 128, 0);
        conv(F(EIx(k, 18)), Wm3f2[k], 128, 128, 0);
        conv(F(EIx(k, 22)), Wm4f1[k], 384, 384, 0);
        conv(F(EIx(k, 24)), Wm4f2[k], 128, 128, 0);
    }
    conv(F(62), Wmsg1[0], 256, 256, 0);  conv(F(64), Wmsg1[1], 256, 256, 0);
    conv(F(66), Wmsg2[0], 128, 128, 0);  conv(F(68), Wmsg2[1], 128, 128, 0);
    conv(F(60), Wr, 128, 256, 0);  conv(F(55), Wr, 128, 256, 128);  // [input_r; hidden_r]
    conv(F(56), Wi, 128, 256, 0);  conv(F(54), Wi, 128, 256, 128);  // [input_i; hidden_i]
    conv(F(58), Wn, 128, 128, 0);
    conv(F(53), Whh, 128, 128, 0);
    conv(F(70), Wo1, 128, 128, 0); conv(F(72), Wo2, 128, 128, 0); conv(F(74), Wo3, 128, 128, 0);
    conv(F(76), Wtr, 128, 128, 0);

    auto bn = [&](float* Xb, int M, const float* g, const float* b) {
        zero_kernel<<<1, 256, 0, stream>>>(stats, 256);
        bnstats_kernel<<<(M + 511) / 512, 128, 0, stream>>>(Xb, stats, M);
        int n = M * HDIM;
        bnapply_kernel<<<(n + 255) / 256, 256, 0, stream>>>(Xb, stats, g, b, M);
    };

    GemmSeg Z{nullptr, nullptr, 0};

    // ---- build flat = transpose(x) ----
    flat_kernel<<<(MBN * TTIME * DDIM + 255) / 256, 256, 0, stream>>>(X, flat);

    // ---- encoder (K=2); nd from last k survives ----
    for (int k = 0; k < 2; ++k) {
        // mlp1
        launch_gemm(ACT_ELU, stream, MBN, 192, GemmSeg{flat, nullptr, 192}, Z, Z, 192, 192,
                    Wm1f1[k], F(EIx(k, 5)), nodeA);
        launch_gemm(ACT_ELU, stream, MBN, 128, GemmSeg{nodeA, nullptr, 128}, Z, Z, 128, 128,
                    Wm1f2[k], F(EIx(k, 7)), nodeB);
        bn(nodeB, MBN, F(EIx(k, 3)), F(EIx(k, 2)));                         // h = nodeB
        // mlp2: concat(h[send], h[recv])
        launch_gemm(ACT_ELU, stream, MBE, 256, GemmSeg{nodeB, send, 128},
                    GemmSeg{nodeB, recv, 128}, Z, 128, 256, Wm2f1[k], F(EIx(k, 11)), EA);
        launch_gemm(ACT_ELU, stream, MBE, 128, GemmSeg{EA, nullptr, 128}, Z, Z, 128, 128,
                    Wm2f2[k], F(EIx(k, 13)), EB);
        bn(EB, MBE, F(EIx(k, 9)), F(EIx(k, 8)));                            // e_skip = EB
        // edge->node aggregation (/N), then mlp3
        agg_kernel<<<(MBN * HDIM + 255) / 256, 256, 0, stream>>>(EB, nodeA, 1.f / NNODE);
        launch_gemm(ACT_ELU, stream, MBN, 128, GemmSeg{nodeA, nullptr, 128}, Z, Z, 128, 128,
                    Wm3f1[k], F(EIx(k, 17)), nodeB);
        launch_gemm(ACT_ELU, stream, MBN, 128, GemmSeg{nodeB, nullptr, 128}, Z, Z, 128, 128,
                    Wm3f2[k], F(EIx(k, 19)), nodeA);
        bn(nodeA, MBN, F(EIx(k, 15)), F(EIx(k, 14)));                       // nd = nodeA
        // mlp4: concat(nd[send], nd[recv], e_skip)
        launch_gemm(ACT_ELU, stream, MBE, 384, GemmSeg{nodeA, send, 128},
                    GemmSeg{nodeA, recv, 128}, GemmSeg{EB, nullptr, 128}, 128, 256,
                    Wm4f1[k], F(EIx(k, 23)), EA);
        launch_gemm(ACT_ELU, stream, MBE, 128, GemmSeg{EA, nullptr, 128}, Z, Z, 128, 128,
                    Wm4f2[k], F(EIx(k, 25)), EA);   // in-place (row-local) is safe
        bn(EA, MBE, F(EIx(k, 21)), F(EIx(k, 20)));
        // logits + softmax over each node's 63 outgoing slots -> rel[:, :, k]
        fcout_kernel<<<(MBE + 255) / 256, 256, 0, stream>>>(EA, F(EIx(k, 0)), F(EIx(k, 1)), logit);
        softmax63_kernel<<<(MBN + 255) / 256, 256, 0, stream>>>(logit, rel, k);
    }

    // ---- decoder init ----
    launch_gemm(ACT_NONE, stream, MBN, 128, GemmSeg{nodeA, nullptr, 128}, Z, Z, 128, 128,
                Wtr, F(77), insB);
    zero_kernel<<<(MBN * HDIM + 255) / 256, 256, 0, stream>>>(hid, MBN * HDIM);
    initacc_kernel<<<(MBN * DDIM + 255) / 256, 256, 0, stream>>>(X, accD);

    // ---- rollout ----
    for (int step = 0; step < RROLL; ++step) {
        for (int k = 0; k < 2; ++k) {
            launch_gemm(ACT_TANH, stream, MBE, 256, GemmSeg{hid, send, 128},
                        GemmSeg{hid, recv, 128}, Z, 128, 256, Wmsg1[k], F(63 + 2 * k), EA);
            // msgs (+)= tanh(EA@W2k + b) * rel[:,:,k] / K
            launch_gemm(ACT_TANH, stream, MBE, 128, GemmSeg{EA, nullptr, 128}, Z, Z, 128, 128,
                        Wmsg2[k], F(67 + 2 * k), EB, nullptr, rel, 2, k, 0.5f, (k == 1));
        }
        agg_kernel<<<(MBN * HDIM + 255) / 256, 256, 0, stream>>>(EB, aggB, 1.f / HDIM);
        // GRU gates: stacked [ins; agg] @ [W_in; W_hidden]
        launch_gemm(ACT_SIGMOID, stream, MBN, 256, GemmSeg{insB, nullptr, 128},
                    GemmSeg{aggB, nullptr, 128}, Z, 128, 256, Wr, F(61), rbuf);
        launch_gemm(ACT_SIGMOID, stream, MBN, 256, GemmSeg{insB, nullptr, 128},
                    GemmSeg{aggB, nullptr, 128}, Z, 128, 256, Wi, F(57), ibuf);
        launch_gemm(ACT_NONE, stream, MBN, 128, GemmSeg{insB, nullptr, 128}, Z, Z, 128, 128,
                    Wn, F(59), n1b);
        launch_gemm(ACT_NONE, stream, MBN, 128, GemmSeg{aggB, nullptr, 128}, Z, Z, 128, 128,
                    Whh, nullptr, n2b);
        gru_kernel<<<(MBN * HDIM + 255) / 256, 256, 0, stream>>>(hid, rbuf, ibuf, n1b, n2b);
        // output MLP; pred = ins + fc3(...) becomes next ins
        launch_gemm(ACT_RELU, stream, MBN, 128, GemmSeg{hid, nullptr, 128}, Z, Z, 128, 128,
                    Wo1, F(71), p1b);
        launch_gemm(ACT_RELU, stream, MBN, 128, GemmSeg{p1b, nullptr, 128}, Z, Z, 128, 128,
                    Wo2, F(73), p2b);
        launch_gemm(ACT_NONE, stream, MBN, 128, GemmSeg{p2b, nullptr, 128}, Z, Z, 128, 128,
                    Wo3, F(75), insB, /*addend=*/insB);
        head_out_kernel<<<(MBN * DDIM + 255) / 256, 256, 0, stream>>>(insB, F(78), F(79),
                                                                      accD, out, step);
    }
}
